// MatchAttention_1511828488808
// MI455X (gfx1250) — compile-verified, matrix operand order checked by reading, not run on hardware
//
#include <hip/hip_runtime.h>
#include <hip/hip_bf16.h>

// ---- problem constants (match reference) ----
#define Bc   2
#define Hc   64
#define Wc   160
#define DIMc 256
#define NHc  8
#define HDc  32
#define Nc   (Hc * Wc)        // 10240
#define BNc  (Bc * Nc)        // 20480
#define Av   16               // window positions
#define SCALEv 0.17677669529663687f  // 32^-0.5

typedef __attribute__((ext_vector_type(16))) _Float16 v16h;
typedef __attribute__((ext_vector_type(4)))  _Float16 v4h;
typedef __attribute__((ext_vector_type(8)))  float    v8f;
typedef _Float16 f16;

// workspace layout (f16 element offsets)
#define WS_QOFF   262144
#define WS_ROWSZ  ((size_t)BNc * 256)

// ------------------------------------------------------------------
// Kernel 0: convert + transpose the four 256x256 fp32 weights to f16
// out[m][n*256 + k] = (f16) W_m[k*256 + n]
// ------------------------------------------------------------------
__global__ __launch_bounds__(256) void prep_weights(const float* __restrict__ wq,
                                                    const float* __restrict__ wk,
                                                    const float* __restrict__ wv,
                                                    const float* __restrict__ wp,
                                                    f16* __restrict__ out) {
    int f = blockIdx.x * 256 + threadIdx.x;       // 0 .. 262143
    int m = f >> 16;
    int r = f & 65535;
    int n = r >> 8;
    int k = r & 255;
    const float* src = (m == 0) ? wq : (m == 1) ? wk : (m == 2) ? wv : wp;
    out[f] = (f16)src[k * 256 + n];
}

// ------------------------------------------------------------------
// Kernel 1: fused QKV GEMM.  32 token rows/block (2 row-tiles), 8 waves.
// Wave w owns 6 of 48 column tiles; B fragments reused across both
// row-tiles.  All fragment base addresses hoisted out of the K loop so
// loads use immediate offsets; B loads software-pipelined 1 deep.
// ------------------------------------------------------------------
__global__ __launch_bounds__(256) void qkv_gemm(const float* __restrict__ x,
                                                const f16* __restrict__ wts,   // 3 transposed weights
                                                f16* __restrict__ qb,
                                                f16* __restrict__ kb,
                                                f16* __restrict__ vb) {
    __shared__ f16 lds[32 * 256];
    const int row0 = blockIdx.x * 32;
    const int tid  = threadIdx.x;

    // stage 32x256 fp32 rows of x into LDS as f16 (float4 in, v4h out)
    #pragma unroll
    for (int i = 0; i < 8; ++i) {
        const int f = (i * 256 + tid) * 4;
        const float4 xv = *(const float4*)&x[(size_t)row0 * 256 + f];
        v4h hv;
        hv[0] = (f16)xv.x; hv[1] = (f16)xv.y; hv[2] = (f16)xv.z; hv[3] = (f16)xv.w;
        *(v4h*)&lds[f] = hv;
    }
    __syncthreads();

    const int wave = tid >> 5;
    const int lane = tid & 31;
    const int half = lane >> 4;
    const int l15  = lane & 15;

    // hoisted fragment base pointers (loop-invariant)
    const f16* bp[6];
    #pragma unroll
    for (int t = 0; t < 6; ++t) {
        const int gt = wave * 6 + t;               // 0..47
        bp[t] = wts + (size_t)(gt >> 4) * 65536
                    + (size_t)((gt & 15) * 16 + l15) * 256 + half * 8;
    }
    const f16* lp0 = &lds[l15 * 256 + half * 8];
    const f16* lp1 = &lds[(16 + l15) * 256 + half * 8];

    auto loadB = [&](int t, int kk) {
        v16h r;
        ((uint4*)&r)[0] = *(const uint4*)(bp[t] + kk * 32);
        ((uint4*)&r)[1] = *(const uint4*)(bp[t] + kk * 32 + 16);
        return r;
    };

    v8f acc[12];                                  // [tile][row-tile]
    #pragma unroll
    for (int t = 0; t < 12; ++t) acc[t] = (v8f){0.f,0.f,0.f,0.f,0.f,0.f,0.f,0.f};

    #pragma unroll
    for (int kk = 0; kk < 8; ++kk) {
        v16h a0, a1;
        ((uint4*)&a0)[0] = *(const uint4*)(lp0 + kk * 32);
        ((uint4*)&a0)[1] = *(const uint4*)(lp0 + kk * 32 + 16);
        ((uint4*)&a1)[0] = *(const uint4*)(lp1 + kk * 32);
        ((uint4*)&a1)[1] = *(const uint4*)(lp1 + kk * 32 + 16);

        v16h bcur = loadB(0, kk);
        #pragma unroll
        for (int t = 0; t < 6; ++t) {
            v16h bnxt = bcur;
            if (t < 5) bnxt = loadB(t + 1, kk);    // prefetch next tile's B
            acc[t * 2 + 0] = __builtin_amdgcn_wmma_f32_16x16x32_f16(
                                 false, a0, false, bcur, (short)0, acc[t * 2 + 0], false, false);
            acc[t * 2 + 1] = __builtin_amdgcn_wmma_f32_16x16x32_f16(
                                 false, a1, false, bcur, (short)0, acc[t * 2 + 1], false, false);
            bcur = bnxt;
        }
    }

    #pragma unroll
    for (int t = 0; t < 6; ++t) {
        const int gt    = wave * 6 + t;
        const int osel  = gt >> 4;
        const int ntile = gt & 15;
        f16* ob = (osel == 0) ? qb : (osel == 1) ? kb : vb;
        const int n = ntile * 16 + l15;
        #pragma unroll
        for (int rt = 0; rt < 2; ++rt) {
            #pragma unroll
            for (int j = 0; j < 8; ++j) {
                const int m = rt * 16 + half * 8 + j;   // C/D VGPR layout
                ob[(size_t)(row0 + m) * 256 + n] = (f16)acc[t * 2 + rt][j];
            }
        }
    }
}

// ------------------------------------------------------------------
// Kernel 2: windowed attention, one wave per token (all 8 heads).
// Lanes (a = lane&15, h = lane>>4) split the 32-dim dot in halves;
// xor-16 shuffle combines, xor-1/2/4/8 reduce softmax over 16 scores.
// ------------------------------------------------------------------
__global__ __launch_bounds__(256) void attn_kernel(const float* __restrict__ moff,
                                                   const f16* __restrict__ qb,
                                                   const f16* __restrict__ kb,
                                                   const f16* __restrict__ vb,
                                                   f16* __restrict__ ao) {
    const int wave = threadIdx.x >> 5;
    const int lane = threadIdx.x & 31;
    const int tok  = blockIdx.x * 8 + wave;
    if (tok >= BNc) return;
    const int bidx = tok / Nc;

    float ox = moff[tok * 2 + 0];
    float oy = moff[tok * 2 + 1];
    ox = fminf(fmaxf(ox, 1.0f), (float)(Wc - 2) - 0.001f);
    oy = fminf(fmaxf(oy, 1.0f), (float)(Hc - 2) - 0.001f);
    const float mx = floorf(ox), my = floorf(oy);
    const float fx = ox - mx,   fy = oy - my;

    const int a     = lane & 15;
    const int halfv = lane >> 4;
    const int axp   = a & 3;
    const int ayp   = a >> 2;
    const int ix    = (int)mx + axp - 1;
    const int iy    = (int)my + ayp - 1;
    const int grow  = bidx * Nc + iy * Wc + ix;    // gather row into k/v

    const float wxv = (axp == 0) ? (1.0f - fx) : (axp == 3) ? fx : 1.0f;
    const float wyv = (ayp == 0) ? (1.0f - fy) : (ayp == 3) ? fy : 1.0f;
    const float bwt = wxv * wyv;

    #pragma unroll
    for (int h = 0; h < NHc; ++h) {
        const f16* qp = qb + (size_t)tok  * 256 + h * 32 + halfv * 16;
        const f16* kp = kb + (size_t)grow * 256 + h * 32 + halfv * 16;
        v16h qv, kv;
        ((uint4*)&qv)[0] = *(const uint4*)qp;
        ((uint4*)&qv)[1] = *(const uint4*)(qp + 8);
        ((uint4*)&kv)[0] = *(const uint4*)kp;
        ((uint4*)&kv)[1] = *(const uint4*)(kp + 8);

        float part = 0.f;
        #pragma unroll
        for (int e = 0; e < 16; ++e)
            part += (float)qv[e] * (float)kv[e];
        part += __shfl_xor(part, 16);              // combine half-dims
        float s = part * SCALEv;

        float mval = s;
        mval = fmaxf(mval, __shfl_xor(mval, 1));
        mval = fmaxf(mval, __shfl_xor(mval, 2));
        mval = fmaxf(mval, __shfl_xor(mval, 4));
        mval = fmaxf(mval, __shfl_xor(mval, 8));

        float ev = __expf(s - mval) * bwt;
        float sum = ev;
        sum += __shfl_xor(sum, 1);
        sum += __shfl_xor(sum, 2);
        sum += __shfl_xor(sum, 4);
        sum += __shfl_xor(sum, 8);
        const float p = ev / sum;

        // accumulate output: lane == dim d (0..31)
        float accd = 0.f;
        #pragma unroll
        for (int aa = 0; aa < Av; ++aa) {
            const float pa = __shfl(p, aa);
            const int   ga = __shfl(grow, aa);
            accd += pa * (float)vb[(size_t)ga * 256 + h * 32 + lane];
        }
        ao[(size_t)tok * 256 + h * 32 + lane] = (f16)accd;
    }
}

// ------------------------------------------------------------------
// Kernel 3: output projection GEMM (f16 in -> f32 out).
// 32 rows/block; wave w owns col-tiles {w, w+8} x row-tiles {0,1}.
// Base pointers hoisted; loads use immediate offsets.
// ------------------------------------------------------------------
__global__ __launch_bounds__(256) void proj_gemm(const f16* __restrict__ ao,
                                                 const f16* __restrict__ wpt,
                                                 float* __restrict__ out) {
    const int row0 = blockIdx.x * 32;
    const int tid  = threadIdx.x;
    const int wave = tid >> 5;
    const int lane = tid & 31;
    const int half = lane >> 4;
    const int l15  = lane & 15;

    const f16* ap0 = ao + (size_t)(row0 + l15) * 256 + half * 8;
    const f16* ap1 = ao + (size_t)(row0 + 16 + l15) * 256 + half * 8;
    const f16* bq0 = wpt + (size_t)(wave * 16 + l15) * 256 + half * 8;
    const f16* bq1 = wpt + (size_t)((wave + 8) * 16 + l15) * 256 + half * 8;

    v8f acc[4];                                    // [col-tile][row-tile]
    #pragma unroll
    for (int t = 0; t < 4; ++t) acc[t] = (v8f){0.f,0.f,0.f,0.f,0.f,0.f,0.f,0.f};

    #pragma unroll
    for (int kk = 0; kk < 8; ++kk) {
        v16h a0, a1, b0, b1;
        ((uint4*)&a0)[0] = *(const uint4*)(ap0 + kk * 32);
        ((uint4*)&a0)[1] = *(const uint4*)(ap0 + kk * 32 + 16);
        ((uint4*)&a1)[0] = *(const uint4*)(ap1 + kk * 32);
        ((uint4*)&a1)[1] = *(const uint4*)(ap1 + kk * 32 + 16);
        ((uint4*)&b0)[0] = *(const uint4*)(bq0 + kk * 32);
        ((uint4*)&b0)[1] = *(const uint4*)(bq0 + kk * 32 + 16);
        ((uint4*)&b1)[0] = *(const uint4*)(bq1 + kk * 32);
        ((uint4*)&b1)[1] = *(const uint4*)(bq1 + kk * 32 + 16);

        acc[0] = __builtin_amdgcn_wmma_f32_16x16x32_f16(false, a0, false, b0, (short)0, acc[0], false, false);
        acc[1] = __builtin_amdgcn_wmma_f32_16x16x32_f16(false, a1, false, b0, (short)0, acc[1], false, false);
        acc[2] = __builtin_amdgcn_wmma_f32_16x16x32_f16(false, a0, false, b1, (short)0, acc[2], false, false);
        acc[3] = __builtin_amdgcn_wmma_f32_16x16x32_f16(false, a1, false, b1, (short)0, acc[3], false, false);
    }

    #pragma unroll
    for (int ct = 0; ct < 2; ++ct) {
        const int ncol = (wave + ct * 8) * 16 + l15;
        #pragma unroll
        for (int rt = 0; rt < 2; ++rt) {
            #pragma unroll
            for (int j = 0; j < 8; ++j) {
                const int m = rt * 16 + half * 8 + j;
                out[(size_t)(row0 + m) * 256 + ncol] = acc[ct * 2 + rt][j];
            }
        }
    }
}

// ------------------------------------------------------------------
extern "C" void kernel_launch(void* const* d_in, const int* in_sizes, int n_in,
                              void* d_out, int out_size, void* d_ws, size_t ws_size,
                              hipStream_t stream) {
    const float* x    = (const float*)d_in[0];  // (B,H,W,DIM)
    const float* moff = (const float*)d_in[1];  // (B,H*W,2)
    const float* Wq   = (const float*)d_in[2];
    const float* Wk   = (const float*)d_in[3];
    const float* Wv   = (const float*)d_in[4];
    const float* Wp   = (const float*)d_in[5];
    float* out = (float*)d_out;

    f16* wsh = (f16*)d_ws;
    f16* wts  = wsh;                            // 3 QKV weights + proj (4*65536)
    f16* wpt  = wsh + 196608;
    f16* qbuf = wsh + WS_QOFF;
    f16* kbuf = qbuf + WS_ROWSZ;
    f16* vbuf = kbuf + WS_ROWSZ;
    f16* aobf = vbuf + WS_ROWSZ;

    prep_weights<<<1024, 256, 0, stream>>>(Wq, Wk, Wv, Wp, wts);
    qkv_gemm<<<BNc / 32, 256, 0, stream>>>(x, wts, qbuf, kbuf, vbuf);
    attn_kernel<<<BNc / 8, 256, 0, stream>>>(moff, qbuf, kbuf, vbuf, aobf);
    proj_gemm<<<BNc / 32, 256, 0, stream>>>(aobf, wpt, out);
}
